// Graph_Encoder_3796751090108
// MI455X (gfx1250) — matrix operand order repaired and checked by reading, not compiled, not run on hardware
//
#include <hip/hip_runtime.h>

// ---------------- problem constants (from reference) ----------------
#define NNODES 100000
#define NEDGES 800000
#define EPLUS  (NEDGES + NNODES)    // edges + self loops
#define NGRAPH 64
#define HEADS  2
#define INCH   128
#define C1CH   64
#define HC1    128                  // HEADS*C1CH
#define C2CH   128
#define HC2    256                  // HEADS*C2CH
#define POOLC  (HC1 + HC2)          // 384
#define BN_EPS 1e-5f
#define NEG_SLOPE 0.2f
#define NEGINF_KEY 0x007FFFFFu      // f2k(-inf)

typedef __attribute__((ext_vector_type(16))) __bf16 v16bf;
typedef __attribute__((ext_vector_type(8)))  float  v8f;
typedef __attribute__((ext_vector_type(4)))  unsigned su4;
typedef __attribute__((ext_vector_type(8)))  unsigned su8;
typedef union { v16bf v; unsigned u[8]; } Frag16;

// ---------------- helpers ----------------
static __device__ __forceinline__ unsigned short f2bf(float f){
  unsigned u = __float_as_uint(f);
  u += 0x7fffu + ((u >> 16) & 1u);          // RNE
  return (unsigned short)(u >> 16);
}
static __device__ __forceinline__ unsigned f2k(float f){   // order-preserving key
  unsigned u = __float_as_uint(f);
  return (u & 0x80000000u) ? ~u : (u | 0x80000000u);
}
static __device__ __forceinline__ float k2f(unsigned k){
  unsigned u = (k & 0x80000000u) ? (k & 0x7fffffffu) : ~k;
  return __uint_as_float(u);
}
static __device__ __forceinline__ float lrelu(float v){
  return v > 0.f ? v : NEG_SLOPE * v;
}
static __device__ __forceinline__ void edge_sd(const int* __restrict__ ei, int e, int& s, int& d){
  if (e < NEDGES){ s = ei[e]; d = ei[NEDGES + e]; }
  else { s = e - NEDGES; d = s; }
}

// ---------------- utility kernels ----------------
__global__ void k_fill(unsigned* __restrict__ p, unsigned v, long long n){
  long long t = (long long)blockIdx.x * blockDim.x + threadIdx.x;
  if (t < n) p[t] = v;
}
__global__ void k_f2bf(const float* __restrict__ x, unsigned short* __restrict__ y, long long n){
  long long t = (long long)blockIdx.x * blockDim.x + threadIdx.x;
  if (t < n) y[t] = f2bf(x[t]);
}

// Pack W[K,M] (f32 row-major) into per-lane B fragments for 16x16x32 bf16 WMMA.
// fragment index = ((ntile*kchunks + kc)*32 + lane)*8 dwords
// lane: col = ntile*16 + lane%16 ; kb = kc*32 + (lane<16?0:16) ; dword j = {K=kb+2j, K=kb+2j+1}
__global__ void k_packW(const float* __restrict__ W, unsigned* __restrict__ frag, int K, int M){
  int t = blockIdx.x * blockDim.x + threadIdx.x;
  int ntiles = M / 16, kchunks = K / 32;
  if (t >= ntiles * kchunks * 32) return;
  int lane = t & 31;
  int wg   = t >> 5;
  int kc   = wg % kchunks;
  int nt   = wg / kchunks;
  int col  = nt * 16 + (lane & 15);
  int kb   = kc * 32 + ((lane < 16) ? 0 : 16);
  unsigned* dst = frag + (size_t)t * 8;
  #pragma unroll
  for (int j = 0; j < 8; ++j){
    unsigned lo = f2bf(W[(size_t)(kb + 2*j    ) * M + col]);
    unsigned hi = f2bf(W[(size_t)(kb + 2*j + 1) * M + col]);
    dst[j] = lo | (hi << 16);
  }
}

// ---------------- WMMA GEMM with TDM + async-LDS staging ----------------
// Block: 256 threads (8 waves). Stages a 96-row A tile via TDM (tensor_load_to_lds)
// and this block's 32KB of packed W fragments via global_load_async_to_lds_b128,
// then computes from LDS with v_wmma_f32_16x16x32_bf16.
template<int K, int M>
__global__ __launch_bounds__(256) void k_gemm_wmma(const unsigned short* __restrict__ A,
                                                   const unsigned* __restrict__ fragB,
                                                   float* __restrict__ C, int nrows){
  constexpr int KC     = K / 32;                 // k chunks (4)
  constexpr int BR     = 96;                     // rows staged per block
  constexpr int RT     = BR / 16;                // 6 row tiles
  constexpr int NTB    = 8;                      // column tiles per block
  constexpr int LDSROW = K * 2 + 16;             // padded row stride (272B): banks shift 4/row
  constexpr int ABYTES = BR * LDSROW;            // 26112
  constexpr int BBYTES = NTB * KC * 32 * 32;     // 32768
  __shared__ __align__(16) char smem[ABYTES + BBYTES];

  const int lane  = threadIdx.x & 31;
  const int wave  = threadIdx.x >> 5;
  const int nHalf = M / (16 * NTB);              // 1 (M=128) or 2 (M=256)
  const int bm    = blockIdx.x / nHalf;
  const int half  = blockIdx.x - bm * nHalf;
  const int rowTiles = nrows / 16;
  const int rt0   = bm * RT;
  int nrt = rowTiles - rt0; if (nrt > RT) nrt = RT;
  if (nrt <= 0) return;                          // block-uniform

  const unsigned ldsA = (unsigned)(unsigned long long)&smem[0];

  // ---- TDM: wave 0 issues one 2D tensor descriptor for the A tile ----
  if (threadIdx.x < 32){
    unsigned long long ga = (unsigned long long)(A + (size_t)rt0 * 16 * K);
    su4 g0; su8 g1;
    g0[0] = 1u;                                                  // count=1, user mode
    g0[1] = ldsA;                                                // lds_addr
    g0[2] = (unsigned)ga;                                        // global_addr[31:0]
    g0[3] = ((unsigned)(ga >> 32) & 0x01FFFFFFu) | 0x80000000u;  // addr[56:32] | type=2
    g1[0] = (1u << 16)            // data_size = 2 bytes
          | (1u << 20)            // pad_enable
          | (5u << 22)            // pad_interval: every 64 dwords (256B)
          | (3u << 25);           // pad_amount: 4 dwords (16B)
    g1[1] = ((unsigned)K & 0xffffu) << 16;                             // tensor_dim0 lo16
    g1[2] = (((unsigned)K >> 16) & 0xffffu)
          | (((unsigned)nrows & 0xffffu) << 16);                       // dim0 hi | dim1 lo
    g1[3] = (((unsigned)nrows >> 16) & 0xffffu)
          | (((unsigned)K & 0xffffu) << 16);                           // dim1 hi | tile_dim0=K
    g1[4] = (unsigned)(nrt * 16);                                      // tile_dim1 rows
    g1[5] = (unsigned)K;                                               // dim0_stride lo32
    g1[6] = 0u;                                                        // stride hi | dim1_stride lo
    g1[7] = 0u;
    asm volatile("tensor_load_to_lds %0, %1" :: "s"(g0), "s"(g1) : "memory");
    __builtin_amdgcn_s_wait_tensorcnt(0);
  }

  // ---- async global->LDS: stage this block's B fragments (32KB) ----
  {
    const unsigned ldsB = ldsA + (unsigned)ABYTES;
    const char* gB = (const char*)fragB + (size_t)half * BBYTES;
    for (int i = threadIdx.x; i < BBYTES / 16; i += 256){
      unsigned la = ldsB + (unsigned)i * 16u;
      unsigned long long gaddr = (unsigned long long)(gB + (size_t)i * 16);
      asm volatile("global_load_async_to_lds_b128 %0, %1, off"
                   :: "v"(la), "v"(gaddr) : "memory");
    }
    asm volatile("s_wait_asynccnt 0x0" ::: "memory");
  }
  __syncthreads();

  // prefetch next row-block's A while computing
  if (threadIdx.x == 0 && rt0 + RT < rowTiles)
    __builtin_prefetch(A + (size_t)(rt0 + RT) * 16 * K, 0, 0);

  // ---- compute from LDS ----
  const char* smA = smem;
  const char* smB = smem + ABYTES;
  const int kb = (lane < 16) ? 0 : 16;           // A-fragment K base (matches packed B)
  for (int t = wave; t < nrt * NTB; t += 8){
    const int mt = t / NTB;
    const int nt = t - mt * NTB;
    const int lr = mt * 16 + (lane & 15);
    v8f acc = {};
    #pragma unroll
    for (int kc = 0; kc < KC; ++kc){
      Frag16 a, b;
      const unsigned short* ap = (const unsigned short*)(smA + lr * LDSROW) + kc * 32 + kb;
      #pragma unroll
      for (int j = 0; j < 8; ++j){
        const int kk = 2*j;                      // VGPR j: K=kb+2j, kb+2j+1
        a.u[j] = *(const unsigned*)(ap + kk);
      }
      const unsigned* bp = (const unsigned*)(smB + ((size_t)((nt * KC + kc) * 32 + lane)) * 32);
      #pragma unroll
      for (int j = 0; j < 8; ++j) b.u[j] = bp[j];
      acc = __builtin_amdgcn_wmma_f32_16x16x32_bf16(false, a.v, false, b.v,
                                                    (short)0, acc, false, false);
    }
    const int col = (half * NTB + nt) * 16 + (lane & 15);
    const int rb  = (rt0 + mt) * 16 + ((lane < 16) ? 0 : 8);
    #pragma unroll
    for (int i = 0; i < 8; ++i)
      C[(size_t)(rb + i) * M + col] = acc[i];    // VGPR i -> rows i / i+8
  }
}

// ---------------- attention node dots: a_s[n,h], a_d[n,h] ----------------
__global__ void k_attdot(const float* __restrict__ h, const float* __restrict__ att_s,
                         const float* __restrict__ att_d, float* __restrict__ so,
                         float* __restrict__ dl, int n, int Cch){
  int t = blockIdx.x * blockDim.x + threadIdx.x;
  if (t >= n * HEADS) return;
  int node = t / HEADS, hd = t % HEADS;
  const float* hp = h + (size_t)node * (HEADS * Cch) + hd * Cch;
  const float* a1 = att_s + hd * Cch;
  const float* a2 = att_d + hd * Cch;
  float s = 0.f, d = 0.f;
  for (int c = 0; c < Cch; ++c){ float v = hp[c]; s += v * a1[c]; d += v * a2[c]; }
  so[t] = s; dl[t] = d;
}

// ---------------- edge softmax passes ----------------
__global__ void k_edge_max(const int* __restrict__ ei, const float* __restrict__ as_,
                           const float* __restrict__ ad_, unsigned* __restrict__ mkey){
  int e = blockIdx.x * blockDim.x + threadIdx.x;
  if (e >= EPLUS) return;
  int s, dn; edge_sd(ei, e, s, dn);
  #pragma unroll
  for (int hd = 0; hd < HEADS; ++hd){
    float a = lrelu(as_[s*HEADS+hd] + ad_[dn*HEADS+hd]);
    atomicMax(&mkey[dn*HEADS+hd], f2k(a));
  }
}
__global__ void k_edge_den(const int* __restrict__ ei, const float* __restrict__ as_,
                           const float* __restrict__ ad_, const unsigned* __restrict__ mkey,
                           float* __restrict__ den){
  int e = blockIdx.x * blockDim.x + threadIdx.x;
  if (e >= EPLUS) return;
  int s, dn; edge_sd(ei, e, s, dn);
  #pragma unroll
  for (int hd = 0; hd < HEADS; ++hd){
    float a = lrelu(as_[s*HEADS+hd] + ad_[dn*HEADS+hd]);
    atomicAdd(&den[dn*HEADS+hd], __expf(a - k2f(mkey[dn*HEADS+hd])));
  }
}
__global__ void k_edge_agg(const int* __restrict__ ei, const float* __restrict__ as_,
                           const float* __restrict__ ad_, const unsigned* __restrict__ mkey,
                           const float* __restrict__ den, const float* __restrict__ hsrc,
                           float* __restrict__ agg, int Cch){
  const int HC = HEADS * Cch;
  const int q4 = HC >> 2;
  long long t = (long long)blockIdx.x * blockDim.x + threadIdx.x;
  if (t >= (long long)EPLUS * q4) return;
  int e = (int)(t / q4);
  int q = (int)(t - (long long)e * q4);
  int s, dn; edge_sd(ei, e, s, dn);
  int hd = (q * 4) / Cch;
  float a = lrelu(as_[s*HEADS+hd] + ad_[dn*HEADS+hd]);
  float w = __expf(a - k2f(mkey[dn*HEADS+hd])) / (den[dn*HEADS+hd] + 1e-16f);
  const float4 hv = *(const float4*)(hsrc + (size_t)s * HC + q * 4);
  float* ap = agg + (size_t)dn * HC + q * 4;
  atomicAdd(ap + 0, hv.x * w); atomicAdd(ap + 1, hv.y * w);
  atomicAdd(ap + 2, hv.z * w); atomicAdd(ap + 3, hv.w * w);
}

// ---------------- batchnorm (train) + relu ----------------
__global__ void k_bn_stats(const float* __restrict__ x, float* __restrict__ gsum,
                           float* __restrict__ gsq, int n){
  int c = threadIdx.x;                 // blockDim.x == C
  int C = blockDim.x;
  int r0 = blockIdx.x * 64;
  int rend = n - r0; if (rend > 64) rend = 64;
  float s = 0.f, q = 0.f;
  for (int i = 0; i < rend; ++i){ float v = x[(size_t)(r0 + i) * C + c]; s += v; q += v * v; }
  atomicAdd(&gsum[c], s); atomicAdd(&gsq[c], q);
}
__global__ void k_bn_final(const float* __restrict__ gsum, const float* __restrict__ gsq,
                           const float* __restrict__ gam, const float* __restrict__ bet,
                           float* __restrict__ scale, float* __restrict__ shift, int n){
  int c = blockIdx.x * blockDim.x + threadIdx.x;
  float mean = gsum[c] / (float)n;
  float var  = gsq[c] / (float)n - mean * mean;
  float sc = gam[c] * rsqrtf(var + BN_EPS);
  scale[c] = sc; shift[c] = bet[c] - mean * sc;
}
template<bool WB>
__global__ void k_bn_apply(const float* __restrict__ x, const float* __restrict__ scale,
                           const float* __restrict__ shift, float* __restrict__ y,
                           unsigned short* __restrict__ yb, int n, int C){
  long long t = (long long)blockIdx.x * blockDim.x + threadIdx.x;
  if (t >= (long long)n * C) return;
  int c = (int)(t % C);
  float v = x[t] * scale[c] + shift[c];
  v = v > 0.f ? v : 0.f;
  y[t] = v;
  if (WB) yb[t] = f2bf(v);
}

// ---------------- pooling ----------------
__global__ void k_count(const int* __restrict__ batch, float* __restrict__ cnt){
  int n = blockIdx.x * blockDim.x + threadIdx.x;
  if (n >= NNODES) return;
  atomicAdd(&cnt[batch[n]], 1.f);
}
__global__ void k_pool(const float* __restrict__ x1, const float* __restrict__ x2,
                       const int* __restrict__ batch, float* __restrict__ psum,
                       unsigned* __restrict__ pmax){
  long long t = (long long)blockIdx.x * blockDim.x + threadIdx.x;
  if (t >= (long long)NNODES * POOLC) return;
  int n = (int)(t / POOLC);
  int c = (int)(t - (long long)n * POOLC);
  float v = (c < HC1) ? x1[(size_t)n * HC1 + c] : x2[(size_t)n * HC2 + (c - HC1)];
  int g = batch[n];
  atomicAdd(&psum[g * POOLC + c], v);
  atomicMax(&pmax[g * POOLC + c], f2k(v));
}
__global__ void k_final(const float* __restrict__ psum, const unsigned* __restrict__ pmax,
                        const float* __restrict__ cnt, float* __restrict__ out){
  int t = blockIdx.x * blockDim.x + threadIdx.x;
  if (t >= NGRAPH * 2 * POOLC) return;
  int g = t / (2 * POOLC);
  int c = t - g * 2 * POOLC;
  out[t] = (c < POOLC) ? psum[g * POOLC + c] / cnt[g]
                       : k2f(pmax[g * POOLC + (c - POOLC)]);
}

// ---------------- host-side orchestration ----------------
static inline unsigned nblk(long long n, int t){ return (unsigned)((n + t - 1) / t); }

extern "C" void kernel_launch(void* const* d_in, const int* in_sizes, int n_in,
                              void* d_out, int out_size, void* d_ws, size_t ws_size,
                              hipStream_t stream){
  const float* x    = (const float*)d_in[0];
  const int*   ei   = (const int*)  d_in[1];
  const int*   bat  = (const int*)  d_in[2];
  const float* W1   = (const float*)d_in[3];
  const float* at_s1= (const float*)d_in[4];
  const float* at_d1= (const float*)d_in[5];
  /* b1 = d_in[6]: cancels inside batchnorm -> unused */
  const float* g1   = (const float*)d_in[7];
  const float* be1  = (const float*)d_in[8];
  const float* W2   = (const float*)d_in[9];
  const float* at_s2= (const float*)d_in[10];
  const float* at_d2= (const float*)d_in[11];
  /* b2 = d_in[12]: unused */
  const float* g2   = (const float*)d_in[13];
  const float* be2  = (const float*)d_in[14];
  float* out = (float*)d_out;

  // ---- workspace layout ----
  char* ws = (char*)d_ws;
  size_t off = 0;
  auto align256 = [](size_t b){ return (b + 255) & ~(size_t)255; };
  auto take = [&](size_t bytes)->char*{ char* p = ws + off; off += align256(bytes); return p; };

  float*          x1   = (float*)          take((size_t)NNODES * HC1 * 4);   // persists
  float*          x2   = (float*)          take((size_t)NNODES * HC2 * 4);   // persists
  unsigned short* x1b  = (unsigned short*) take((size_t)NNODES * HC1 * 2);   // GEMM2 input
  // scratch region shared between layers
  const size_t xbB  = align256((size_t)NNODES * INCH * 2);
  const size_t h1B  = align256((size_t)NNODES * HC1  * 4);
  const size_t h2B  = align256((size_t)NNODES * HC2  * 4);
  const size_t s1sz = xbB + 2 * h1B;          // xb + h1 + agg1
  const size_t s2sz = 2 * h2B;                // h2 + agg2
  char* S = take(s1sz > s2sz ? s1sz : s2sz);
  unsigned short* xb   = (unsigned short*)S;
  float*          h1   = (float*)(S + xbB);
  float*          agg1 = (float*)(S + xbB + h1B);
  float*          h2   = (float*)S;
  float*          agg2 = (float*)(S + h2B);
  // small buffers
  unsigned* fragW1 = (unsigned*) take((size_t)INCH * HC1 * 2);
  unsigned* fragW2 = (unsigned*) take((size_t)HC1  * HC2 * 2);
  float*    asb  = (float*)    take((size_t)NNODES * HEADS * 4);
  float*    adb  = (float*)    take((size_t)NNODES * HEADS * 4);
  unsigned* mkey = (unsigned*) take((size_t)NNODES * HEADS * 4);
  float*    den  = (float*)    take((size_t)NNODES * HEADS * 4);
  float*    gsum = (float*)    take(256 * 4);
  float*    gsq  = (float*)    take(256 * 4);
  float*    bsc  = (float*)    take(256 * 4);
  float*    bsh  = (float*)    take(256 * 4);
  float*    psum = (float*)    take((size_t)NGRAPH * POOLC * 4);
  unsigned* pmax = (unsigned*) take((size_t)NGRAPH * POOLC * 4);
  float*    cnt  = (float*)    take((size_t)NGRAPH * 4);

  const int T = 256;
  const long long NH = (long long)NNODES * HEADS;
  const int rowTiles = NNODES / 16;                  // 6250
  const unsigned gemmBlocks = (unsigned)((rowTiles + 5) / 6);  // BR=96 -> RT=6

  // ===== layer 1 =====
  k_f2bf<<<nblk((long long)NNODES * INCH, T), T, 0, stream>>>(x, xb, (long long)NNODES * INCH);
  k_packW<<<nblk((INCH/32) * (HC1/16) * 32, T), T, 0, stream>>>(W1, fragW1, INCH, HC1);
  k_gemm_wmma<INCH, HC1><<<gemmBlocks * (HC1/128), T, 0, stream>>>(xb, fragW1, h1, NNODES);
  k_attdot<<<nblk(NH, T), T, 0, stream>>>(h1, at_s1, at_d1, asb, adb, NNODES, C1CH);
  k_fill<<<nblk(NH, T), T, 0, stream>>>(mkey, NEGINF_KEY, NH);
  k_fill<<<nblk(NH, T), T, 0, stream>>>((unsigned*)den, 0u, NH);
  k_fill<<<nblk((long long)NNODES * HC1, T), T, 0, stream>>>((unsigned*)agg1, 0u, (long long)NNODES * HC1);
  k_edge_max<<<nblk(EPLUS, T), T, 0, stream>>>(ei, asb, adb, mkey);
  k_edge_den<<<nblk(EPLUS, T), T, 0, stream>>>(ei, asb, adb, mkey, den);
  k_edge_agg<<<nblk((long long)EPLUS * (HC1/4), T), T, 0, stream>>>(ei, asb, adb, mkey, den, h1, agg1, C1CH);
  k_fill<<<1, 256, 0, stream>>>((unsigned*)gsum, 0u, 256);
  k_fill<<<1, 256, 0, stream>>>((unsigned*)gsq,  0u, 256);
  k_bn_stats<<<nblk(NNODES, 64), HC1, 0, stream>>>(agg1, gsum, gsq, NNODES);
  k_bn_final<<<1, HC1, 0, stream>>>(gsum, gsq, g1, be1, bsc, bsh, NNODES);
  k_bn_apply<true><<<nblk((long long)NNODES * HC1, T), T, 0, stream>>>(agg1, bsc, bsh, x1, x1b, NNODES, HC1);

  // ===== layer 2 =====
  k_packW<<<nblk((HC1/32) * (HC2/16) * 32, T), T, 0, stream>>>(W2, fragW2, HC1, HC2);
  k_gemm_wmma<HC1, HC2><<<gemmBlocks * (HC2/128), T, 0, stream>>>(x1b, fragW2, h2, NNODES);
  k_attdot<<<nblk(NH, T), T, 0, stream>>>(h2, at_s2, at_d2, asb, adb, NNODES, C2CH);
  k_fill<<<nblk(NH, T), T, 0, stream>>>(mkey, NEGINF_KEY, NH);
  k_fill<<<nblk(NH, T), T, 0, stream>>>((unsigned*)den, 0u, NH);
  k_fill<<<nblk((long long)NNODES * HC2, T), T, 0, stream>>>((unsigned*)agg2, 0u, (long long)NNODES * HC2);
  k_edge_max<<<nblk(EPLUS, T), T, 0, stream>>>(ei, asb, adb, mkey);
  k_edge_den<<<nblk(EPLUS, T), T, 0, stream>>>(ei, asb, adb, mkey, den);
  k_edge_agg<<<nblk((long long)EPLUS * (HC2/4), T), T, 0, stream>>>(ei, asb, adb, mkey, den, h2, agg2, C2CH);
  k_fill<<<1, 256, 0, stream>>>((unsigned*)gsum, 0u, 256);
  k_fill<<<1, 256, 0, stream>>>((unsigned*)gsq,  0u, 256);
  k_bn_stats<<<nblk(NNODES, 64), HC2, 0, stream>>>(agg2, gsum, gsq, NNODES);
  k_bn_final<<<1, HC2, 0, stream>>>(gsum, gsq, g2, be2, bsc, bsh, NNODES);
  k_bn_apply<false><<<nblk((long long)NNODES * HC2, T), T, 0, stream>>>(agg2, bsc, bsh, x2, nullptr, NNODES, HC2);

  // ===== pooling =====
  k_fill<<<nblk((long long)NGRAPH * POOLC, T), T, 0, stream>>>((unsigned*)psum, 0u, (long long)NGRAPH * POOLC);
  k_fill<<<nblk((long long)NGRAPH * POOLC, T), T, 0, stream>>>(pmax, NEGINF_KEY, (long long)NGRAPH * POOLC);
  k_fill<<<1, 64, 0, stream>>>((unsigned*)cnt, 0u, NGRAPH);
  k_count<<<nblk(NNODES, T), T, 0, stream>>>(bat, cnt);
  k_pool<<<nblk((long long)NNODES * POOLC, T), T, 0, stream>>>(x1, x2, bat, psum, pmax);
  k_final<<<nblk(NGRAPH * 2 * POOLC, T), T, 0, stream>>>(psum, pmax, cnt, out);
}
